// LRU_18159121727720
// MI455X (gfx1250) — matrix operand order repaired and checked by reading, not compiled, not run on hardware
//
#include <hip/hip_runtime.h>
#include <hip/hip_bf16.h>
#include <stdint.h>

// ---------------------------------------------------------------------------
// LRU single step on MI455X (gfx1250), wave32 + WMMA bf16, double-buffered LDS.
//   GEMM1: Bu (re|im stacked, N=8192) fused with lambda-mix + gamma epilogue
//   GEMM2: y = hidden_re@C_re.T - hidden_im@C_im.T + D*inputs  (K=8192 stacked)
// Workspace: planar hidden (re then im), 2*2048*4096 floats = 64 MB.
// d_out: y (4M floats) followed by hidden as interleaved complex64 (re,im).
// ---------------------------------------------------------------------------

typedef __bf16 v16bf __attribute__((ext_vector_type(16)));
typedef __bf16 v2bf  __attribute__((ext_vector_type(2)));
typedef float  v8f   __attribute__((ext_vector_type(8)));
typedef float  v2f   __attribute__((ext_vector_type(2)));

constexpr int BSZ  = 2048;   // batch
constexpr int DM   = 2048;   // d_model
constexpr int H2   = 4096;   // 2*H (hidden complex width)
constexpr int HD   = 2048;   // H per 2x2 block
constexpr int NF   = 8192;   // fused N for GEMM1 (re|im)
constexpr int KF   = 8192;   // fused K for GEMM2 (re|im)

constexpr int BM  = 128;
constexpr int BN  = 128;
constexpr int BK  = 32;
constexpr int LDT = 40;      // halves per LDS row: 32 data + 8 pad (80B, conflict-free)

union Frag { uint4 q[2]; v16bf v; };

// Packed RNE float2 -> bf16x2: lowers directly to v_cvt_pk_bf16_f32.
__device__ __forceinline__ unsigned int pk_bf16(float lo, float hi) {
    v2f f = {lo, hi};
    v2bf p = __builtin_convertvector(f, v2bf);
    unsigned int u;
    __builtin_memcpy(&u, &p, 4);
    return u;
}

// 128x32 f32 tile staged by 256 threads: 4 float4 each, held in registers
// across the compute phase (double-buffer software pipeline).
struct StageRegs { float4 v[4]; };

__device__ __forceinline__ void load_regs(const float* __restrict__ g, int ldg,
                                          int tid, StageRegs& r, bool prefetch_next)
{
#pragma unroll
    for (int it = 0; it < 4; ++it) {
        int q   = it * 256 + tid;            // 0..1023
        int row = q >> 3;                    // 0..127
        int col = (q & 7) << 2;              // 0,4,...,28
        const float* p = g + (size_t)row * ldg + col;
        r.v[it] = *(const float4*)p;
        if (it == 0 && prefetch_next) __builtin_prefetch(p + BK, 0, 1);  // K+2 tile
    }
}

__device__ __forceinline__ void store_regs(const StageRegs& r, unsigned short* t,
                                           int tid, bool neg)
{
#pragma unroll
    for (int it = 0; it < 4; ++it) {
        int q   = it * 256 + tid;
        int row = q >> 3;
        int col = (q & 7) << 2;
        float4 v = r.v[it];
        if (neg) { v.x = -v.x; v.y = -v.y; v.z = -v.z; v.w = -v.w; }
        *(uint2*)&t[row * LDT + col] = make_uint2(pk_bf16(v.x, v.y), pk_bf16(v.z, v.w));
    }
}

// One K-step of the 128x128 block: each wave does its 64x32 sub-tile
// = 8 x v_wmma_f32_16x16x32_bf16.
// A frag (16x32 bf16): lane<16 -> K 0..15      ; lane>=16 -> K 16..31
// B frag (32x16 bf16): lane<16 -> K 0..15      ; lane>=16 -> K 16..31
__device__ __forceinline__ void mma_step(const unsigned short* At,
                                         const unsigned short* Bt,
                                         int waveM, int waveN, int l16, int half,
                                         v8f (&acc)[4][2])
{
    Frag b0, b1;
    const int kbB = half ? 16 : 0;
    {
        int r0 = (waveN * 32 + l16) * LDT + kbB;
        b0.q[0] = *(const uint4*)&Bt[r0];
        b0.q[1] = *(const uint4*)&Bt[r0 + 8];
        int r1 = (waveN * 32 + 16 + l16) * LDT + kbB;
        b1.q[0] = *(const uint4*)&Bt[r1];
        b1.q[1] = *(const uint4*)&Bt[r1 + 8];
    }
    const int kbA = half ? 16 : 0;
#pragma unroll
    for (int mi = 0; mi < 4; ++mi) {
        Frag a;
        int ra = (waveM * 64 + mi * 16 + l16) * LDT + kbA;
        a.q[0] = *(const uint4*)&At[ra];
        a.q[1] = *(const uint4*)&At[ra + 8];
        acc[mi][0] = __builtin_amdgcn_wmma_f32_16x16x32_bf16(
            false, a.v, false, b0.v, (short)0, acc[mi][0], false, false);
        acc[mi][1] = __builtin_amdgcn_wmma_f32_16x16x32_bf16(
            false, a.v, false, b1.v, (short)0, acc[mi][1], false, false);
    }
}

// ---------------------------------------------------------------------------
// GEMM1: raw = inputs @ [B_re|B_im].T   (M=2048, N=8192, K=2048)
// epilogue: hidden = 0.5 * (lam 2x2-mix of h_old) + exp(gamma_log) * raw
// ---------------------------------------------------------------------------
__global__ __launch_bounds__(256) void lru_gemm1(
    const float* __restrict__ inputs,
    const float* __restrict__ Bre, const float* __restrict__ Bim,
    const float* __restrict__ h_re, const float* __restrict__ h_im,
    const float* __restrict__ nu,  const float* __restrict__ theta,
    const float* __restrict__ gamma_log,
    float* __restrict__ wsRe, float* __restrict__ wsIm,
    float* __restrict__ hidOut /* interleaved, length 2*BSZ*H2 */)
{
    __shared__ __align__(16) unsigned short At[2][BM * LDT];
    __shared__ __align__(16) unsigned short Bt[2][BN * LDT];

    const int tid    = threadIdx.x;
    const int nStart = blockIdx.x * BN;          // 0..NF-1, never straddles 4096
    const int mStart = blockIdx.y * BM;
    const bool planeIm = (nStart >= H2);

    const float* Asrc = inputs + (size_t)mStart * DM;
    const float* Bsrc = planeIm ? (Bim + (size_t)(nStart - H2) * DM)
                                : (Bre + (size_t)nStart * DM);

    const int wave = tid >> 5, lane = tid & 31;
    const int l16 = lane & 15, half = lane >> 4;
    const int waveM = wave >> 2, waveN = wave & 3;

    v8f acc[4][2];
#pragma unroll
    for (int mi = 0; mi < 4; ++mi)
#pragma unroll
        for (int ni = 0; ni < 2; ++ni) acc[mi][ni] = {};

    // software pipeline: prime buffer 0
    StageRegs rA, rB;
    load_regs(Asrc, DM, tid, rA, true);
    load_regs(Bsrc, DM, tid, rB, true);
    store_regs(rA, At[0], tid, false);
    store_regs(rB, Bt[0], tid, false);

    const int KT = DM / BK;
    int cur = 0;
    for (int kt = 0; kt < KT; ++kt) {
        const bool more = (kt + 1 < KT);
        if (more) {
            load_regs(Asrc + (kt + 1) * BK, DM, tid, rA, kt + 2 < KT);
            load_regs(Bsrc + (kt + 1) * BK, DM, tid, rB, kt + 2 < KT);
        }
        __syncthreads();                       // buf[cur] fully staged
        mma_step(At[cur], Bt[cur], waveM, waveN, l16, half, acc);
        if (more) {
            store_regs(rA, At[cur ^ 1], tid, false);
            store_regs(rB, Bt[cur ^ 1], tid, false);
        }
        cur ^= 1;
    }

    // epilogue: per-column lambda/gamma hoisted, per-row h_old mix
#pragma unroll
    for (int ni = 0; ni < 2; ++ni) {
        const int ng  = nStart + waveN * 32 + ni * 16 + l16;  // 0..8191
        const int col = ng & (H2 - 1);                        // 0..4095
        const int i2  = col >> 11;                            // block row (Hd=2048)
        const int h   = col & (HD - 1);
        const float g = __expf(gamma_log[col]);
        float lr[2], li[2];
#pragma unroll
        for (int j = 0; j < 2; ++j) {
            const int idx = ((i2 << 1) | j) * HD + h;
            const float mag = __expf(-__expf(nu[idx]));
            const float ang = __expf(theta[idx]);
            lr[j] = mag * __cosf(ang);
            li[j] = mag * __sinf(ang);
        }
#pragma unroll
        for (int mi = 0; mi < 4; ++mi) {
#pragma unroll
            for (int r = 0; r < 8; ++r) {
                const int bg = mStart + waveM * 64 + mi * 16 + r + (half ? 8 : 0);
                const size_t hb = (size_t)bg * H2;
                const float hr0 = h_re[hb + h],      hi0 = h_im[hb + h];
                const float hr1 = h_re[hb + HD + h], hi1 = h_im[hb + HD + h];
                const float ls = planeIm
                    ? (lr[0] * hi0 + li[0] * hr0 + lr[1] * hi1 + li[1] * hr1)
                    : (lr[0] * hr0 - li[0] * hi0 + lr[1] * hr1 - li[1] * hi1);
                const float out = 0.5f * ls + g * acc[mi][ni][r];
                const size_t off = hb + col;
                (planeIm ? wsIm : wsRe)[off] = out;              // planar for GEMM2
                hidOut[off * 2 + (planeIm ? 1 : 0)] = out;       // complex64 out
            }
        }
    }
}

// ---------------------------------------------------------------------------
// GEMM2: y = [hidden_re|hidden_im] @ [C_re|-C_im].T + D*inputs
//        (M=2048, N=2048, K=8192; K-tiles 0..127 from re, 128..255 from im)
// ---------------------------------------------------------------------------
__global__ __launch_bounds__(256) void lru_gemm2(
    const float* __restrict__ wsRe, const float* __restrict__ wsIm,
    const float* __restrict__ Cre,  const float* __restrict__ Cim,
    const float* __restrict__ Dv,   const float* __restrict__ inputs,
    float* __restrict__ y)
{
    __shared__ __align__(16) unsigned short At[2][BM * LDT];
    __shared__ __align__(16) unsigned short Bt[2][BN * LDT];

    const int tid    = threadIdx.x;
    const int nStart = blockIdx.x * BN;
    const int mStart = blockIdx.y * BM;

    const int wave = tid >> 5, lane = tid & 31;
    const int l16 = lane & 15, half = lane >> 4;
    const int waveM = wave >> 2, waveN = wave & 3;

    v8f acc[4][2];
#pragma unroll
    for (int mi = 0; mi < 4; ++mi)
#pragma unroll
        for (int ni = 0; ni < 2; ++ni) acc[mi][ni] = {};

    auto aPtr = [&](int kt) {
        int k0 = kt * BK; bool im = (k0 >= H2); int kk = k0 & (H2 - 1);
        return (im ? wsIm : wsRe) + (size_t)mStart * H2 + kk;
    };
    auto bPtr = [&](int kt) {
        int k0 = kt * BK; bool im = (k0 >= H2); int kk = k0 & (H2 - 1);
        return (im ? Cim : Cre) + (size_t)nStart * H2 + kk;
    };

    StageRegs rA, rB;
    load_regs(aPtr(0), H2, tid, rA, true);
    load_regs(bPtr(0), H2, tid, rB, true);
    store_regs(rA, At[0], tid, false);
    store_regs(rB, Bt[0], tid, false);

    const int KT = KF / BK;
    int cur = 0;
    for (int kt = 0; kt < KT; ++kt) {
        const bool more    = (kt + 1 < KT);
        const bool negNext = ((kt + 1) * BK >= H2);   // fold -C_im into convert
        if (more) {
            load_regs(aPtr(kt + 1), H2, tid, rA, kt + 2 < KT);
            load_regs(bPtr(kt + 1), H2, tid, rB, kt + 2 < KT);
        }
        __syncthreads();
        mma_step(At[cur], Bt[cur], waveM, waveN, l16, half, acc);
        if (more) {
            store_regs(rA, At[cur ^ 1], tid, false);
            store_regs(rB, Bt[cur ^ 1], tid, negNext);
        }
        cur ^= 1;
    }

#pragma unroll
    for (int ni = 0; ni < 2; ++ni) {
        const int ng = nStart + waveN * 32 + ni * 16 + l16;   // 0..2047
        const float dn = Dv[ng];
#pragma unroll
        for (int mi = 0; mi < 4; ++mi) {
#pragma unroll
            for (int r = 0; r < 8; ++r) {
                const int bg = mStart + waveM * 64 + mi * 16 + r + (half ? 8 : 0);
                const size_t off = (size_t)bg * DM + ng;
                y[off] = acc[mi][ni][r] + dn * inputs[off];
            }
        }
    }
}

// ---------------------------------------------------------------------------
extern "C" void kernel_launch(void* const* d_in, const int* in_sizes, int n_in,
                              void* d_out, int out_size, void* d_ws, size_t ws_size,
                              hipStream_t stream) {
    (void)in_sizes; (void)n_in; (void)out_size; (void)ws_size;
    const float* inputs    = (const float*)d_in[0];
    const float* h_re      = (const float*)d_in[1];
    const float* h_im      = (const float*)d_in[2];
    const float* nu        = (const float*)d_in[3];
    const float* theta     = (const float*)d_in[4];
    const float* gamma_log = (const float*)d_in[5];
    const float* B_re      = (const float*)d_in[6];
    const float* B_im      = (const float*)d_in[7];
    const float* C_re      = (const float*)d_in[8];
    const float* C_im      = (const float*)d_in[9];
    const float* Dv        = (const float*)d_in[10];

    float* y      = (float*)d_out;
    float* hidOut = y + (size_t)BSZ * DM;              // interleaved complex64
    float* wsRe   = (float*)d_ws;                      // planar hidden_re
    float* wsIm   = wsRe + (size_t)BSZ * H2;           // planar hidden_im

    dim3 blk(256);
    lru_gemm1<<<dim3(NF / BN, BSZ / BM), blk, 0, stream>>>(
        inputs, B_re, B_im, h_re, h_im, nu, theta, gamma_log, wsRe, wsIm, hidOut);
    lru_gemm2<<<dim3(DM / BN, BSZ / BM), blk, 0, stream>>>(
        wsRe, wsIm, C_re, C_im, Dv, inputs, y);
}
